// PointSetAbstraction_3865470566873
// MI455X (gfx1250) — matrix operand order repaired and checked by reading, not compiled, not run on hardware
//
#include <hip/hip_runtime.h>
#include <hip/hip_bf16.h>

typedef __attribute__((ext_vector_type(16))) _Float16 v16h;
typedef __attribute__((ext_vector_type(8)))  _Float16 v8h;
typedef __attribute__((ext_vector_type(8)))  float    v8f;
typedef __attribute__((ext_vector_type(4)))  int      v4i;

#if defined(__has_builtin)
#if __has_builtin(__builtin_amdgcn_global_load_async_to_lds_b128)
#define HAVE_ASYNC_COPY 1
#endif
#if __has_builtin(__builtin_amdgcn_s_wait_asynccnt)
#define HAVE_WAIT_ASYNC 1
#endif
#endif

// Problem constants (match reference setup_inputs)
constexpr int Bb   = 8;
constexpr int Nn   = 4096;
constexpr int CIN  = 64;
constexpr int Ss   = 1024;   // NPOINT
constexpr int Kk   = 16;     // K neighbors
constexpr int COUT = 128;
constexpr int C0   = 67;     // CIN+3
constexpr int C0P  = 96;     // padded to mult of 32
constexpr int EC   = 134;
constexpr int ECP  = 160;    // padded to mult of 32 (and 16 for N)
constexpr int C3P  = 128;
constexpr long long R = (long long)Bb * Ss * Kk;  // 131072 rows

// ---------------------------------------------------------------------------
// 1) Farthest point sampling: one block per batch. Writes centroid indices and
//    new_xyz (first chunk of d_out).
// ---------------------------------------------------------------------------
__global__ __launch_bounds__(256) void fps_kernel(const float* __restrict__ xyz,
                                                  int* __restrict__ cent,
                                                  float* __restrict__ newxyz) {
  __shared__ float dist[Nn];
  __shared__ float rv[256];
  __shared__ int   ri[256];
  __shared__ int   sFar;
  const int b = blockIdx.x;
  const int t = threadIdx.x;
  const float* px = xyz + (long long)b * Nn * 3;
  for (int i = t; i < Nn; i += 256) dist[i] = 1e10f;
  if (t == 0) sFar = 0;
  __syncthreads();
  for (int it = 0; it < Ss; ++it) {
    const int far = sFar;
    const float cx = px[far * 3 + 0];
    const float cy = px[far * 3 + 1];
    const float cz = px[far * 3 + 2];
    if (t == 0) {
      cent[b * Ss + it] = far;
      float* o = newxyz + ((long long)b * Ss + it) * 3;
      o[0] = cx; o[1] = cy; o[2] = cz;
    }
    float bv = -1.0f; int bi = 0;
    for (int i = t; i < Nn; i += 256) {
      const float dx = px[i * 3 + 0] - cx;
      const float dy = px[i * 3 + 1] - cy;
      const float dz = px[i * 3 + 2] - cz;
      const float d  = dx * dx + dy * dy + dz * dz;
      const float nd = fminf(dist[i], d);
      dist[i] = nd;
      if (nd > bv) { bv = nd; bi = i; }
    }
    rv[t] = bv; ri[t] = bi;
    __syncthreads();
    for (int s2 = 128; s2 > 0; s2 >>= 1) {
      if (t < s2) {
        if (rv[t + s2] > rv[t] || (rv[t + s2] == rv[t] && ri[t + s2] < ri[t])) {
          rv[t] = rv[t + s2]; ri[t] = ri[t + s2];
        }
      }
      __syncthreads();
    }
    if (t == 0) sFar = ri[0];
    __syncthreads();
  }
}

// ---------------------------------------------------------------------------
// 2) kNN: one thread per query, candidate points tiled in LDS (48 KB of the
//    320 KB WGP LDS). Insertion list of 16 (sorted ascending, like top_k(-d)).
// ---------------------------------------------------------------------------
__global__ __launch_bounds__(128) void knn_kernel(const float* __restrict__ xyz,
                                                  const float* __restrict__ newxyz,
                                                  int* __restrict__ knn) {
  __shared__ float spx[Nn], spy[Nn], spz[Nn];
  const int b = blockIdx.y;
  const int q = blockIdx.x * 128 + threadIdx.x;
  const float* px = xyz + (long long)b * Nn * 3;
  for (int i = threadIdx.x; i < Nn; i += 128) {
    spx[i] = px[i * 3 + 0];
    spy[i] = px[i * 3 + 1];
    spz[i] = px[i * 3 + 2];
  }
  __syncthreads();
  const float* qp = newxyz + ((long long)b * Ss + q) * 3;
  const float qx = qp[0], qy = qp[1], qz = qp[2];
  float bd[Kk]; int bi[Kk];
#pragma unroll
  for (int j = 0; j < Kk; ++j) { bd[j] = 3.4e38f; bi[j] = 0; }
  for (int n = 0; n < Nn; ++n) {
    const float dx = spx[n] - qx, dy = spy[n] - qy, dz = spz[n] - qz;
    const float d = dx * dx + dy * dy + dz * dz;
    if (d < bd[Kk - 1]) {
      int j = Kk - 1;
      while (j > 0 && bd[j - 1] > d) { bd[j] = bd[j - 1]; bi[j] = bi[j - 1]; --j; }
      bd[j] = d; bi[j] = n;
    }
  }
  int* o = knn + ((long long)(b * Ss + q)) * Kk;
  for (int j = 0; j < Kk; ++j) o[j] = bi[j];
}

// ---------------------------------------------------------------------------
// 3) Build X = [grouped_xyz - new_xyz, grouped_pts] as f16, padded 67 -> 96.
// ---------------------------------------------------------------------------
__global__ __launch_bounds__(256) void build_x_kernel(const float* __restrict__ xyz,
                                                      const float* __restrict__ pts,
                                                      const float* __restrict__ newxyz,
                                                      const int* __restrict__ knn,
                                                      _Float16* __restrict__ X) {
  const long long r = (long long)blockIdx.x * 256 + threadIdx.x;
  if (r >= R) return;
  const long long bs = r / Kk;
  const int b = (int)(bs / Ss);
  const int i = knn[r];
  const float* p = xyz + ((long long)b * Nn + i) * 3;
  const float* q = newxyz + bs * 3;
  _Float16* xr = X + r * C0P;
  xr[0] = (_Float16)(p[0] - q[0]);
  xr[1] = (_Float16)(p[1] - q[1]);
  xr[2] = (_Float16)(p[2] - q[2]);
  const float* f = pts + ((long long)b * Nn + i) * CIN;
  for (int c = 0; c < CIN; ++c) xr[3 + c] = (_Float16)f[c];
  for (int c = C0; c < C0P; ++c) xr[c] = (_Float16)0.0f;
}

// ---------------------------------------------------------------------------
// 4) Pack an fp32 weight matrix [Kin, Nin] (row-major) into the per-lane
//    WMMA B-fragment layout, zero-padded to [KP, NP].
//    B 32x16 f16 striping: lane half selects K range (0..15 / 16..31),
//    lane&15 selects the N column, element j walks K within the half.
//    Packed index: ((kt*NT + nt)*32 + lane)*16 + j with
//    k = kt*32 + (lane>>4)*16 + j, n = nt*16 + (lane&15).
// ---------------------------------------------------------------------------
__global__ void pack_w_kernel(const float* __restrict__ W, _Float16* __restrict__ Wp,
                              int Kin, int Nin, int KP, int NP) {
  const int NT = NP / 16;
  const long long total = (long long)(KP / 32) * NT * 512;
  const long long e = (long long)blockIdx.x * 256 + threadIdx.x;
  if (e >= total) return;
  const int j = (int)(e & 15);
  const int lane = (int)((e >> 4) & 31);
  const int tile = (int)(e >> 9);
  const int nt = tile % NT;
  const int kt = tile / NT;
  const int k = kt * 32 + (lane >> 4) * 16 + j;
  const int n = nt * 16 + (lane & 15);
  const float v = (k < Kin && n < Nin) ? W[(long long)k * Nin + n] : 0.0f;
  Wp[e] = (_Float16)v;
}

// ---------------------------------------------------------------------------
// 5) WMMA GEMM: D[R, NP] = A[R, KP] @ B[KP, NP], f16 storage, f32 accumulate.
//    The whole packed-B panel (24-50 KB) is staged into LDS once per block via
//    async global->LDS copies (ASYNCcnt path) and reused by all 8 waves.
//    One wave per 16-row M-tile covering all N-tiles.
//    A 16x32 f16 operand: lanes 0-15 hold K {0..15}, lanes 16-31 hold
//    K {16..31} (contiguous, as packed B) -> two 16B loads fused with shufflevector.
// ---------------------------------------------------------------------------
template <int KP, int NP>
__global__ __launch_bounds__(256) void gemm_f16_wmma(const _Float16* __restrict__ A,
                                                     const _Float16* __restrict__ Bp,
                                                     _Float16* __restrict__ D) {
  constexpr int KT = KP / 32;
  constexpr int NT = NP / 16;
  constexpr int BELEMS = KT * NT * 512;   // f16 elements in packed B panel
  __shared__ _Float16 Bs[BELEMS];

  // Cooperative copy of the packed-B panel into LDS (16B per lane per chunk).
  for (int ch = threadIdx.x; ch < BELEMS / 8; ch += 256) {
#if HAVE_ASYNC_COPY
    __builtin_amdgcn_global_load_async_to_lds_b128(
        (__attribute__((address_space(1))) v4i*)(Bp + ch * 8),
        (__attribute__((address_space(3))) v4i*)(&Bs[ch * 8]), 0, 0);
#else
    *(v8h*)(&Bs[ch * 8]) = *(const v8h*)(Bp + ch * 8);
#endif
  }
#if HAVE_ASYNC_COPY
#if HAVE_WAIT_ASYNC
  __builtin_amdgcn_s_wait_asynccnt(0);
#else
  asm volatile("s_wait_asynccnt 0x0" ::: "memory");
#endif
#endif
  __syncthreads();

  const int wave = threadIdx.x >> 5;
  const int lane = threadIdx.x & 31;
  const int m0 = (blockIdx.x * 8 + wave) * 16;
  const int arow = m0 + (lane & 15);
  const int kbase = (lane >> 4) * 16;     // contiguous K per lane half (matches packed B)

  v8f zero = {};
  v8f acc[NT];
#pragma unroll
  for (int nt = 0; nt < NT; ++nt) acc[nt] = zero;

#pragma unroll
  for (int kt = 0; kt < KT; ++kt) {
    const _Float16* ap = A + (long long)arow * KP + kt * 32 + kbase;
    v8h alo = *(const v8h*)(ap);
    v8h ahi = *(const v8h*)(ap + 8);
    v16h a = __builtin_shufflevector(alo, ahi, 0, 1, 2, 3, 4, 5, 6, 7,
                                     8, 9, 10, 11, 12, 13, 14, 15);
#pragma unroll
    for (int nt = 0; nt < NT; ++nt) {
      v16h b = *(const v16h*)(&Bs[((kt * NT + nt) * 32 + lane) << 4]);
      acc[nt] = __builtin_amdgcn_wmma_f32_16x16x32_f16(
          /*neg_a=*/false, a, /*neg_b=*/false, b,
          /*c_mod=*/(short)0, acc[nt], /*reuse_a=*/false, /*reuse_b=*/false);
    }
  }

  const int half = lane >> 4;
  const int col  = lane & 15;
#pragma unroll
  for (int nt = 0; nt < NT; ++nt) {
#pragma unroll
    for (int v = 0; v < 8; ++v) {
      const int row = m0 + half * 8 + v;
      D[(long long)row * NP + nt * 16 + col] = (_Float16)acc[nt][v];
    }
  }
}

// ---------------------------------------------------------------------------
// 6) BN statistics: per-channel sum & sumsq. blockDim = NP (mult of 32).
// ---------------------------------------------------------------------------
__global__ void stats_kernel(const _Float16* __restrict__ Xf, float* __restrict__ acc,
                             int NP, int rowsPerBlock) {
  const int c = threadIdx.x;
  const long long r0 = (long long)blockIdx.x * rowsPerBlock;
  float s = 0.0f, sq = 0.0f;
  for (int rl = 0; rl < rowsPerBlock; ++rl) {
    const float x = (float)Xf[(r0 + rl) * NP + c];
    s += x; sq += x * x;
  }
  atomicAdd(&acc[c], s);
  atomicAdd(&acc[NP + c], sq);
}

__global__ void zero_kernel(float* p, int n) {
  const int i = blockIdx.x * 256 + threadIdx.x;
  if (i < n) p[i] = 0.0f;
}

// scale/shift from accumulated stats; pad channels forced to 0 so downstream
// elementwise kernels keep pads zeroed.
__global__ void finalize_stats_kernel(const float* __restrict__ acc,
                                      const float* __restrict__ gamma,
                                      const float* __restrict__ beta,
                                      float* __restrict__ scale,
                                      float* __restrict__ shift,
                                      int NP, int Cvalid, float invR) {
  const int c = threadIdx.x;
  if (c >= NP) return;
  if (c >= Cvalid) { scale[c] = 0.0f; shift[c] = 0.0f; return; }
  const float m = acc[c] * invR;
  const float v = acc[NP + c] * invR - m * m;
  const float sc = gamma[c] * rsqrtf(v + 1e-5f);
  scale[c] = sc;
  shift[c] = beta[c] - m * sc;
}

// In-place BN + ReLU (layers 1 and 2).
__global__ __launch_bounds__(256) void bn_apply_kernel(_Float16* __restrict__ Xf,
                                                       const float* __restrict__ scale,
                                                       const float* __restrict__ shift,
                                                       long long total, int NP) {
  const long long i = (long long)blockIdx.x * 256 + threadIdx.x;
  if (i >= total) return;
  const int c = (int)(i % NP);
  float y = (float)Xf[i] * scale[c] + shift[c];
  y = fmaxf(y, 0.0f);
  Xf[i] = (_Float16)y;
}

// ---------------------------------------------------------------------------
// 7) Final fuse: relu(bn3(h3) + bnSC(sc)), then max over K neighbors.
// ---------------------------------------------------------------------------
__global__ __launch_bounds__(256) void final_kernel(const _Float16* __restrict__ H3,
                                                    const _Float16* __restrict__ SC,
                                                    const float* __restrict__ s3,
                                                    const float* __restrict__ t3,
                                                    const float* __restrict__ sS,
                                                    const float* __restrict__ tS,
                                                    float* __restrict__ outPts) {
  const long long i = (long long)blockIdx.x * 256 + threadIdx.x;
  if (i >= (long long)Bb * Ss * COUT) return;
  const int c = (int)(i % COUT);
  const long long bs = i / COUT;
  const float a3 = s3[c], b3 = t3[c], aS = sS[c], bS = tS[c];
  float m = -3.4e38f;
#pragma unroll
  for (int k = 0; k < Kk; ++k) {
    const long long r = bs * Kk + k;
    float v = (float)H3[r * C3P + c] * a3 + b3 + (float)SC[r * C3P + c] * aS + bS;
    v = fmaxf(v, 0.0f);
    m = fmaxf(m, v);
  }
  outPts[i] = m;
}

// ---------------------------------------------------------------------------
// Launch orchestration
// ---------------------------------------------------------------------------
static inline size_t alignup(size_t x) { return (x + 255) & ~(size_t)255; }

extern "C" void kernel_launch(void* const* d_in, const int* in_sizes, int n_in,
                              void* d_out, int out_size, void* d_ws, size_t ws_size,
                              hipStream_t stream) {
  const float* xyz    = (const float*)d_in[0];
  const float* pts    = (const float*)d_in[1];
  const float* fc1_w  = (const float*)d_in[2];
  const float* bn1_g  = (const float*)d_in[4];
  const float* bn1_b  = (const float*)d_in[5];
  const float* fc2_w  = (const float*)d_in[6];
  const float* bn2_g  = (const float*)d_in[8];
  const float* bn2_b  = (const float*)d_in[9];
  const float* fc3_w  = (const float*)d_in[10];
  const float* bn3_g  = (const float*)d_in[12];
  const float* bn3_b  = (const float*)d_in[13];
  const float* sc_w   = (const float*)d_in[14];
  const float* scbn_g = (const float*)d_in[16];
  const float* scbn_b = (const float*)d_in[17];

  float* out       = (float*)d_out;
  float* out_xyz   = out;                          // [B,S,3]
  float* out_pts   = out + (long long)Bb * Ss * 3; // [B,S,COUT]

  // Workspace layout
  char* ws = (char*)d_ws;
  size_t off = 0;
  auto take = [&](size_t bytes) { size_t o = off; off = alignup(off + bytes); return o; };
  int*       cent  = (int*)      (ws + take((size_t)Bb * Ss * 4));
  int*       knn   = (int*)      (ws + take((size_t)Bb * Ss * Kk * 4));
  _Float16*  Wp1   = (_Float16*) (ws + take((size_t)(C0P/32)*(ECP/16)*512*2));
  _Float16*  Wp2   = (_Float16*) (ws + take((size_t)(ECP/32)*(ECP/16)*512*2));
  _Float16*  Wp3   = (_Float16*) (ws + take((size_t)(ECP/32)*(C3P/16)*512*2));
  _Float16*  WpS   = (_Float16*) (ws + take((size_t)(C0P/32)*(C3P/16)*512*2));
  float*     accA  = (float*)    (ws + take((size_t)(2*ECP + 2*ECP + 2*C3P + 2*C3P) * 4));
  float*     acc1  = accA;               // 2*160
  float*     acc2  = accA + 2*ECP;       // 2*160
  float*     acc3  = accA + 4*ECP;       // 2*128
  float*     accS  = accA + 4*ECP + 2*C3P;
  float*     ssA   = (float*)    (ws + take((size_t)(2*ECP + 2*ECP + 2*C3P + 2*C3P) * 4));
  float*     sc1 = ssA,            *sh1 = ssA + ECP;
  float*     sc2 = ssA + 2*ECP,    *sh2 = ssA + 3*ECP;
  float*     sc3 = ssA + 4*ECP,    *sh3 = ssA + 4*ECP + C3P;
  float*     scS = ssA + 4*ECP + 2*C3P, *shS = ssA + 4*ECP + 3*C3P;
  _Float16*  Xb    = (_Float16*) (ws + take((size_t)R * C0P * 2));
  _Float16*  Hb    = (_Float16*) (ws + take((size_t)R * ECP * 2));
  _Float16*  H2b   = (_Float16*) (ws + take((size_t)R * ECP * 2));
  _Float16*  SCb   = (_Float16*) (ws + take((size_t)R * C3P * 2));
  (void)ws_size; (void)in_sizes; (void)n_in; (void)out_size;

  const int accN = 4 * ECP + 4 * C3P; // total accumulator floats
  const float invR = 1.0f / (float)R;

  // --- Stage 1: FPS + kNN + grouping ---
  fps_kernel<<<Bb, 256, 0, stream>>>(xyz, cent, out_xyz);
  knn_kernel<<<dim3(Ss / 128, Bb), 128, 0, stream>>>(xyz, out_xyz, knn);
  build_x_kernel<<<(int)((R + 255) / 256), 256, 0, stream>>>(xyz, pts, out_xyz, knn, Xb);

  // --- Weight packing (biases cancel through train-mode BN) ---
  auto packBlocks = [](int KP, int NP) { return (int)(((size_t)(KP/32)*(NP/16)*512 + 255) / 256); };
  pack_w_kernel<<<packBlocks(C0P, ECP), 256, 0, stream>>>(fc1_w, Wp1, C0, EC, C0P, ECP);
  pack_w_kernel<<<packBlocks(ECP, ECP), 256, 0, stream>>>(fc2_w, Wp2, EC, EC, ECP, ECP);
  pack_w_kernel<<<packBlocks(ECP, C3P), 256, 0, stream>>>(fc3_w, Wp3, EC, COUT, ECP, C3P);
  pack_w_kernel<<<packBlocks(C0P, C3P), 256, 0, stream>>>(sc_w,  WpS, C0, COUT, C0P, C3P);
  zero_kernel<<<(accN + 255) / 256, 256, 0, stream>>>(accA, accN);

  const int gemmBlocks = (int)(R / 128);   // 8 waves x 16 rows per block
  const int statBlocks = (int)(R / 1024);

  // --- h1 = X @ W1 ; bn1 ; relu (in place) ---
  gemm_f16_wmma<C0P, ECP><<<gemmBlocks, 256, 0, stream>>>(Xb, Wp1, Hb);
  stats_kernel<<<statBlocks, ECP, 0, stream>>>(Hb, acc1, ECP, 1024);
  finalize_stats_kernel<<<1, ECP, 0, stream>>>(acc1, bn1_g, bn1_b, sc1, sh1, ECP, EC, invR);
  bn_apply_kernel<<<(int)((R * ECP + 255) / 256), 256, 0, stream>>>(Hb, sc1, sh1, R * ECP, ECP);

  // --- shortcut sc = X @ Wsc ; stats only (BN fused in final) ---
  gemm_f16_wmma<C0P, C3P><<<gemmBlocks, 256, 0, stream>>>(Xb, WpS, SCb);
  stats_kernel<<<statBlocks, C3P, 0, stream>>>(SCb, accS, C3P, 1024);
  finalize_stats_kernel<<<1, C3P, 0, stream>>>(accS, scbn_g, scbn_b, scS, shS, C3P, COUT, invR);

  // --- h2 = a1 @ W2 ; bn2 ; relu (in place) ---
  gemm_f16_wmma<ECP, ECP><<<gemmBlocks, 256, 0, stream>>>(Hb, Wp2, H2b);
  stats_kernel<<<statBlocks, ECP, 0, stream>>>(H2b, acc2, ECP, 1024);
  finalize_stats_kernel<<<1, ECP, 0, stream>>>(acc2, bn2_g, bn2_b, sc2, sh2, ECP, EC, invR);
  bn_apply_kernel<<<(int)((R * ECP + 255) / 256), 256, 0, stream>>>(H2b, sc2, sh2, R * ECP, ECP);

  // --- h3 = a2 @ W3 (into Hb, a1 dead) ; stats (BN fused in final) ---
  gemm_f16_wmma<ECP, C3P><<<gemmBlocks, 256, 0, stream>>>(H2b, Wp3, Hb);
  stats_kernel<<<statBlocks, C3P, 0, stream>>>(Hb, acc3, C3P, 1024);
  finalize_stats_kernel<<<1, C3P, 0, stream>>>(acc3, bn3_g, bn3_b, sc3, sh3, C3P, COUT, invR);

  // --- relu(bn3(h3)+bnSC(sc)) , max over K ---
  final_kernel<<<(int)(((long long)Bb * Ss * COUT + 255) / 256), 256, 0, stream>>>(
      Hb, SCb, sc3, sh3, scS, shS, out_pts);
}
